// Attention_29326036697594
// MI455X (gfx1250) — compile-verified, matrix operand order checked by reading, not run on hardware
//
#include <hip/hip_runtime.h>
#include <hip/hip_bf16.h>

// MI455X / gfx1250: wave32, WMMA 16x16x32 bf16 with f32 accumulate.
// Compute-bound problem (~412 GFLOP vs ~0.65 GB of mandatory HBM traffic),
// so all three GEMMs run on v_wmma_f32_16x16x32_bf16 with explicit
// double-buffered K-loops (issue next fragment loads before waiting on the
// current set) and precomputed per-lane base pointers so addresses fold into
// instruction offsets.

typedef __bf16 bf16;
typedef __attribute__((ext_vector_type(16))) __bf16 v16bf;
typedef __attribute__((ext_vector_type(8)))  __bf16 v8bf;
typedef __attribute__((ext_vector_type(8)))  float  v8f;

#define BB 16
#define TT 2048
#define SS 2048
#define HH 1024

static __device__ __forceinline__ v8f wmma_bf16(v16bf a, v16bf b, v8f c) {
    return __builtin_amdgcn_wmma_f32_16x16x32_bf16(
        /*neg_a=*/false, a, /*neg_b=*/false, b,
        /*c_mod=*/(short)0, c, /*reuse_a=*/false, /*reuse_b=*/false);
}

// 16-bit A-matrix 16x32 fragment (B uses the same per-lane pattern, lane=N):
// lane L (L<16):  row M=L,    elems 0-7 = K[0..7],  elems 8-15 = K[16..23]
// lane L (>=16):  row M=L-16, elems 0-7 = K[8..15], elems 8-15 = K[24..31]
// p = per-lane pointer: base + (mn0 + (lane&15))*ld + (lane>>4)*8 + k
static __device__ __forceinline__ v16bf frag_at(const bf16* __restrict__ p) {
    v8bf lo = *(const v8bf*)(p);
    v8bf hi = *(const v8bf*)(p + 16);
    v16bf f;
#pragma unroll
    for (int i = 0; i < 8; ++i) { f[i] = lo[i]; f[i + 8] = hi[i]; }
    return f;
}

// Same fragment, f32 source converted on the fly (used for the softmax
// probability matrix so a bf16 copy is never materialized).
static __device__ __forceinline__ v16bf frag_at_f32(const float* __restrict__ p) {
    v8f lo = *(const v8f*)(p);
    v8f hi = *(const v8f*)(p + 16);
    v16bf f;
#pragma unroll
    for (int i = 0; i < 8; ++i) { f[i] = (bf16)lo[i]; f[i + 8] = (bf16)hi[i]; }
    return f;
}

// ---------------------------------------------------------------- conversions
__global__ __launch_bounds__(256) void cvt_f32_to_bf16(const float* __restrict__ in,
                                                       bf16* __restrict__ out, size_t n) {
    size_t i = (size_t)blockIdx.x * blockDim.x + threadIdx.x;
    size_t stride = (size_t)gridDim.x * blockDim.x;
    for (; i < n; i += stride) out[i] = (bf16)in[i];
}

// enc: f32 [B,S,H] -> bf16 row-major copy AND bf16 transposed copy [B,H,S]
// (transposed copy gives contiguous-K B-fragments for the context GEMM).
__global__ __launch_bounds__(256) void enc_cvt_transpose(const float* __restrict__ enc,
                                                         bf16* __restrict__ enc_bf,
                                                         bf16* __restrict__ encT_bf) {
    __shared__ bf16 tile[32][33];
    const int b  = blockIdx.z;
    const int s0 = blockIdx.y * 32;
    const int h0 = blockIdx.x * 32;
    const float* src = enc + ((size_t)b * SS + s0) * HH + h0;
#pragma unroll
    for (int i = threadIdx.y; i < 32; i += 8) {
        bf16 bv = (bf16)src[(size_t)i * HH + threadIdx.x];
        tile[i][threadIdx.x] = bv;
        enc_bf[((size_t)b * SS + s0 + i) * HH + h0 + threadIdx.x] = bv;
    }
    __syncthreads();
    bf16* dst = encT_bf + ((size_t)b * HH + h0) * SS + s0;
#pragma unroll
    for (int i = threadIdx.y; i < 32; i += 8)
        dst[(size_t)i * SS + threadIdx.x] = tile[threadIdx.x][i];
}

// ---------------------------------------------------------------- scores GEMM
// attn[b,t,s] = sum_h dec[b,t,h]*enc[b,s,h].  Block tile 128x128, 8 waves,
// each wave 64x32 (4x2 WMMA accumulators), K-step 32, double-buffered.
__global__ __launch_bounds__(256) void scores_kernel(const bf16* __restrict__ dec_bf,
                                                     const bf16* __restrict__ enc_bf,
                                                     float* __restrict__ attn) {
    const int b      = blockIdx.z;
    const int tile_t = blockIdx.y * 128;
    const int tile_s = blockIdx.x * 128;
    const int wave   = threadIdx.x >> 5;
    const int lane   = threadIdx.x & 31;
    const int wrow   = (wave & 1) * 64;   // 2 wave rows  -> 128 M
    const int wcol   = (wave >> 1) * 32;  // 4 wave cols  -> 128 N
    const int lrow   = lane & 15;
    const int lhalf  = (lane >> 4) * 8;

    const bf16* A  = dec_bf + (size_t)b * TT * HH;
    const bf16* Bm = enc_bf + (size_t)b * SS * HH;
    float*      C  = attn   + (size_t)b * TT * SS;

    // Per-lane base pointers (K offset added in-loop; row offsets constant).
    const bf16* pA = A  + (size_t)(tile_t + wrow + lrow) * HH + lhalf;
    const bf16* pB = Bm + (size_t)(tile_s + wcol + lrow) * HH + lhalf;

    v8f acc[4][2];
#pragma unroll
    for (int i = 0; i < 4; ++i)
#pragma unroll
        for (int j = 0; j < 2; ++j) acc[i][j] = (v8f)0.0f;

    v16bf af[2][4], bfr[2][2];
#pragma unroll
    for (int i = 0; i < 4; ++i) af[0][i] = frag_at(pA + i * 16 * HH);
#pragma unroll
    for (int j = 0; j < 2; ++j) bfr[0][j] = frag_at(pB + j * 16 * HH);

    for (int k0 = 0; k0 < HH; k0 += 64) {
        // issue loads for set 1 (k0+32) before computing set 0
#pragma unroll
        for (int i = 0; i < 4; ++i) af[1][i] = frag_at(pA + i * 16 * HH + k0 + 32);
#pragma unroll
        for (int j = 0; j < 2; ++j) bfr[1][j] = frag_at(pB + j * 16 * HH + k0 + 32);
#pragma unroll
        for (int i = 0; i < 4; ++i)
#pragma unroll
            for (int j = 0; j < 2; ++j)
                acc[i][j] = wmma_bf16(af[0][i], bfr[0][j], acc[i][j]);
        if (k0 + 64 < HH) {
#pragma unroll
            for (int i = 0; i < 4; ++i) af[0][i] = frag_at(pA + i * 16 * HH + k0 + 64);
#pragma unroll
            for (int j = 0; j < 2; ++j) bfr[0][j] = frag_at(pB + j * 16 * HH + k0 + 64);
        }
#pragma unroll
        for (int i = 0; i < 4; ++i)
#pragma unroll
            for (int j = 0; j < 2; ++j)
                acc[i][j] = wmma_bf16(af[1][i], bfr[1][j], acc[i][j]);
    }

    const int crow  = tile_t + wrow + (lane >> 4) * 8;
    const int ccol0 = tile_s + wcol + (lane & 15);
#pragma unroll
    for (int i = 0; i < 4; ++i)
#pragma unroll
        for (int j = 0; j < 2; ++j)
#pragma unroll
            for (int r = 0; r < 8; ++r)
                C[(size_t)(crow + 16 * i + r) * SS + ccol0 + 16 * j] = acc[i][j][r];
}

// ---------------------------------------------------------------- softmax
// One block (8 waves) per row of 2048; wave32 shfl + LDS reduction; in place.
__global__ __launch_bounds__(256) void softmax_kernel(float* __restrict__ attn) {
    float* p = attn + (size_t)blockIdx.x * SS;
    const int tid  = threadIdx.x;
    const int lane = tid & 31;
    const int wave = tid >> 5;
    __shared__ float red[8];

    float v[8];
#pragma unroll
    for (int i = 0; i < 8; ++i) v[i] = p[tid + 256 * i];

    float m = v[0];
#pragma unroll
    for (int i = 1; i < 8; ++i) m = fmaxf(m, v[i]);
#pragma unroll
    for (int off = 16; off > 0; off >>= 1) m = fmaxf(m, __shfl_xor(m, off, 32));
    if (lane == 0) red[wave] = m;
    __syncthreads();
    float mall = red[0];
#pragma unroll
    for (int w = 1; w < 8; ++w) mall = fmaxf(mall, red[w]);
    __syncthreads();

    float s = 0.0f;
#pragma unroll
    for (int i = 0; i < 8; ++i) { v[i] = __expf(v[i] - mall); s += v[i]; }
#pragma unroll
    for (int off = 16; off > 0; off >>= 1) s += __shfl_xor(s, off, 32);
    if (lane == 0) red[wave] = s;
    __syncthreads();
    float tot = 0.0f;
#pragma unroll
    for (int w = 0; w < 8; ++w) tot += red[w];
    const float inv = 1.0f / tot;
#pragma unroll
    for (int i = 0; i < 8; ++i) p[tid + 256 * i] = v[i] * inv;
}

// ---------------------------------------------------------------- context GEMM
// ctx[b,t,h] = sum_s P[b,t,s]*enc[b,s,h].  A = f32 probs (cvt on the fly;
// per-batch P is 16 MB -> L2-resident), B columns = rows of transposed
// encoder (contiguous K). Output bf16. Double-buffered K-loop.
__global__ __launch_bounds__(256) void context_kernel(const float* __restrict__ attn,
                                                      const bf16* __restrict__ encT_bf,
                                                      bf16* __restrict__ ctx_bf) {
    const int b      = blockIdx.z;
    const int tile_t = blockIdx.y * 128;
    const int tile_h = blockIdx.x * 128;
    const int wave   = threadIdx.x >> 5;
    const int lane   = threadIdx.x & 31;
    const int wrow   = (wave & 1) * 64;
    const int wcol   = (wave >> 1) * 32;
    const int lrow   = lane & 15;
    const int lhalf  = (lane >> 4) * 8;

    const float* A  = attn    + (size_t)b * TT * SS;
    const bf16*  Bm = encT_bf + (size_t)b * HH * SS;
    bf16*        C  = ctx_bf  + (size_t)b * TT * HH;

    const float* pA = A  + (size_t)(tile_t + wrow + lrow) * SS + lhalf;
    const bf16*  pB = Bm + (size_t)(tile_h + wcol + lrow) * SS + lhalf;

    v8f acc[4][2];
#pragma unroll
    for (int i = 0; i < 4; ++i)
#pragma unroll
        for (int j = 0; j < 2; ++j) acc[i][j] = (v8f)0.0f;

    v16bf af[2][4], bfr[2][2];
#pragma unroll
    for (int i = 0; i < 4; ++i) af[0][i] = frag_at_f32(pA + i * 16 * SS);
#pragma unroll
    for (int j = 0; j < 2; ++j) bfr[0][j] = frag_at(pB + j * 16 * SS);

    for (int k0 = 0; k0 < SS; k0 += 64) {
#pragma unroll
        for (int i = 0; i < 4; ++i) af[1][i] = frag_at_f32(pA + i * 16 * SS + k0 + 32);
#pragma unroll
        for (int j = 0; j < 2; ++j) bfr[1][j] = frag_at(pB + j * 16 * SS + k0 + 32);
#pragma unroll
        for (int i = 0; i < 4; ++i)
#pragma unroll
            for (int j = 0; j < 2; ++j)
                acc[i][j] = wmma_bf16(af[0][i], bfr[0][j], acc[i][j]);
        if (k0 + 64 < SS) {
#pragma unroll
            for (int i = 0; i < 4; ++i) af[0][i] = frag_at_f32(pA + i * 16 * SS + k0 + 64);
#pragma unroll
            for (int j = 0; j < 2; ++j) bfr[0][j] = frag_at(pB + j * 16 * SS + k0 + 64);
        }
#pragma unroll
        for (int i = 0; i < 4; ++i)
#pragma unroll
            for (int j = 0; j < 2; ++j)
                acc[i][j] = wmma_bf16(af[1][i], bfr[1][j], acc[i][j]);
    }

    const int crow  = tile_t + wrow + (lane >> 4) * 8;
    const int ccol0 = tile_h + wcol + (lane & 15);
#pragma unroll
    for (int i = 0; i < 4; ++i)
#pragma unroll
        for (int j = 0; j < 2; ++j)
#pragma unroll
            for (int r = 0; r < 8; ++r)
                C[(size_t)(crow + 16 * i + r) * HH + ccol0 + 16 * j] =
                    (bf16)acc[i][j][r];
}

// ---------------------------------------------------------------- projection
// out[b,t,h] = tanh( sum_{k<H} ctx[t,k]*W[h,k] + sum_{k<H} dec[t,k]*W[h,H+k]
//                    + bias[h] ).
// Concat handled by per-k pointer select (64-step divides the H boundary).
// W rows are the B columns directly (row-major, contiguous K); W is 4 MB bf16
// -> fully L2-resident across all 2048 blocks.
__global__ __launch_bounds__(256) void proj_kernel(const bf16* __restrict__ ctx_bf,
                                                   const bf16* __restrict__ dec_bf,
                                                   const bf16* __restrict__ W_bf,
                                                   const float* __restrict__ bias,
                                                   float* __restrict__ out) {
    const int b      = blockIdx.z;
    const int tile_t = blockIdx.y * 128;
    const int tile_h = blockIdx.x * 128;
    const int wave   = threadIdx.x >> 5;
    const int lane   = threadIdx.x & 31;
    const int wrow   = (wave & 1) * 64;
    const int wcol   = (wave >> 1) * 32;
    const int lrow   = lane & 15;
    const int lhalf  = (lane >> 4) * 8;

    const bf16* pA1 = ctx_bf + (size_t)b * TT * HH
                      + (size_t)(tile_t + wrow + lrow) * HH + lhalf;   // K in [0,H)
    const bf16* pA2 = dec_bf + (size_t)b * TT * HH
                      + (size_t)(tile_t + wrow + lrow) * HH + lhalf;   // K in [H,2H)
    const bf16* pB  = W_bf + (size_t)(tile_h + wcol + lrow) * (2 * HH) + lhalf;
    float*      C   = out  + (size_t)b * TT * HH;

    v8f acc[4][2];
#pragma unroll
    for (int i = 0; i < 4; ++i)
#pragma unroll
        for (int j = 0; j < 2; ++j) acc[i][j] = (v8f)0.0f;

    v16bf af[2][4], bfr[2][2];
#pragma unroll
    for (int i = 0; i < 4; ++i) af[0][i] = frag_at(pA1 + i * 16 * HH);
#pragma unroll
    for (int j = 0; j < 2; ++j) bfr[0][j] = frag_at(pB + j * 16 * (2 * HH));

    for (int k0 = 0; k0 < 2 * HH; k0 += 64) {
        const int k1 = k0 + 32;
        const bf16* a1 = (k1 < HH) ? (pA1 + k1) : (pA2 + (k1 - HH));
#pragma unroll
        for (int i = 0; i < 4; ++i) af[1][i] = frag_at(a1 + i * 16 * HH);
#pragma unroll
        for (int j = 0; j < 2; ++j) bfr[1][j] = frag_at(pB + j * 16 * (2 * HH) + k1);
#pragma unroll
        for (int i = 0; i < 4; ++i)
#pragma unroll
            for (int j = 0; j < 2; ++j)
                acc[i][j] = wmma_bf16(af[0][i], bfr[0][j], acc[i][j]);
        if (k0 + 64 < 2 * HH) {
            const int k2 = k0 + 64;
            const bf16* a0 = (k2 < HH) ? (pA1 + k2) : (pA2 + (k2 - HH));
#pragma unroll
            for (int i = 0; i < 4; ++i) af[0][i] = frag_at(a0 + i * 16 * HH);
#pragma unroll
            for (int j = 0; j < 2; ++j) bfr[0][j] = frag_at(pB + j * 16 * (2 * HH) + k2);
        }
#pragma unroll
        for (int i = 0; i < 4; ++i)
#pragma unroll
            for (int j = 0; j < 2; ++j)
                acc[i][j] = wmma_bf16(af[1][i], bfr[1][j], acc[i][j]);
    }

    const int crow  = tile_t + wrow + (lane >> 4) * 8;
    const int ccol0 = tile_h + wcol + (lane & 15);
#pragma unroll
    for (int i = 0; i < 4; ++i)
#pragma unroll
        for (int j = 0; j < 2; ++j) {
            const int   col = ccol0 + 16 * j;
            const float bv  = bias[col];
#pragma unroll
            for (int r = 0; r < 8; ++r)
                C[(size_t)(crow + 16 * i + r) * HH + col] =
                    tanhf(acc[i][j][r] + bv);
        }
}

// ---------------------------------------------------------------- launch
extern "C" void kernel_launch(void* const* d_in, const int* in_sizes, int n_in,
                              void* d_out, int out_size, void* d_ws, size_t ws_size,
                              hipStream_t stream) {
    (void)in_sizes; (void)n_in; (void)out_size; (void)ws_size;

    const float* enc  = (const float*)d_in[0];  // [B,S,H]
    const float* dec  = (const float*)d_in[1];  // [B,T,H]
    const float* W    = (const float*)d_in[2];  // [H,2H]
    const float* bias = (const float*)d_in[3];  // [H]

    float* out  = (float*)d_out;                          // [B,T,H] f32
    float* attn = out + (size_t)BB * TT * HH;             // [B,T,S] f32

    // Workspace layout (bf16 buffers, ~260 MB total):
    const size_t n_bsh = (size_t)BB * SS * HH;            // 33.5M elems
    bf16* enc_bf  = (bf16*)d_ws;                          // [B,S,H]
    bf16* encT_bf = enc_bf  + n_bsh;                      // [B,H,S]
    bf16* dec_bf  = encT_bf + n_bsh;                      // [B,T,H]
    bf16* W_bf    = dec_bf  + n_bsh;                      // [H,2H]
    bf16* ctx_bf  = W_bf    + (size_t)HH * 2 * HH;        // [B,T,H]

    // 1) Precision conversions (memory-bound, ~20us at 23.3 TB/s).
    cvt_f32_to_bf16<<<8192, 256, 0, stream>>>(dec, dec_bf, n_bsh);
    cvt_f32_to_bf16<<<2048, 256, 0, stream>>>(W, W_bf, (size_t)HH * 2 * HH);
    enc_cvt_transpose<<<dim3(HH / 32, SS / 32, BB), dim3(32, 8), 0, stream>>>(
        enc, enc_bf, encT_bf);

    // 2) scores = D * E^T           (137 GFLOP, bf16 WMMA)
    scores_kernel<<<dim3(SS / 128, TT / 128, BB), 256, 0, stream>>>(
        dec_bf, enc_bf, attn);

    // 3) row softmax, in place in d_out's attn region
    softmax_kernel<<<dim3(BB * TT), 256, 0, stream>>>(attn);

    // 4) context = P * E            (137 GFLOP, bf16 WMMA, A cvt'd on the fly)
    context_kernel<<<dim3(HH / 128, TT / 128, BB), 256, 0, stream>>>(
        attn, encT_bf, ctx_bf);

    // 5) out = tanh([ctx, dec] * W^T + b)   (137 GFLOP, bf16 WMMA)
    proj_kernel<<<dim3(HH / 128, TT / 128, BB), 256, 0, stream>>>(
        ctx_bf, dec_bf, W_bf, bias, out);
}
